// VectorQuantizer_51951924412440
// MI455X (gfx1250) — matrix operand order repaired and checked by reading, not compiled, not run on hardware
//
#include <hip/hip_runtime.h>
#include <stdint.h>

#define EMB_NUM 1024
#define EMB_DIM 256
#define N_ROWS  (64 * 32 * 32)      // 65536 flattened z rows
#define ROWS_PER_WG 128
#define THREADS 256
#define NUM_WG (N_ROWS / ROWS_PER_WG)  // 512

typedef __attribute__((ext_vector_type(16))) __bf16 v16bf;
typedef __attribute__((ext_vector_type(8)))  float  v8f;

union BF16Frag { v16bf v; uint32_t u[8]; };

__device__ __forceinline__ uint32_t bf16_rne(float x) {
  uint32_t u = __float_as_uint(x);
  return (u + 0x7FFFu + ((u >> 16) & 1u)) >> 16;
}
__device__ __forceinline__ uint32_t pack_bf16(float lo, float hi) {
  return (bf16_rne(hi) << 16) | (bf16_rne(lo) & 0xFFFFu);
}

// ---------------------------------------------------------------------------
// Kernel 1: codebook prep.  One wave per codebook row: emit packed-bf16 copy
// (layout: row-major, 128 uint32 per row = 256 bf16) and fp32 ||e||^2.
// ---------------------------------------------------------------------------
__global__ __launch_bounds__(256) void vq_prep(const float* __restrict__ emb,
                                               uint32_t* __restrict__ emb_bf,
                                               float* __restrict__ enorm) {
  const int wave = (blockIdx.x * blockDim.x + threadIdx.x) >> 5;  // row id
  const int lane = threadIdx.x & 31;
  if (wave >= EMB_NUM) return;

  const float4* row4 = (const float4*)(emb + wave * EMB_DIM);
  float4 f0 = row4[lane * 2 + 0];
  float4 f1 = row4[lane * 2 + 1];

  float s = f0.x * f0.x + f0.y * f0.y + f0.z * f0.z + f0.w * f0.w +
            f1.x * f1.x + f1.y * f1.y + f1.z * f1.z + f1.w * f1.w;

  uint4 pk;
  pk.x = pack_bf16(f0.x, f0.y);
  pk.y = pack_bf16(f0.z, f0.w);
  pk.z = pack_bf16(f1.x, f1.y);
  pk.w = pack_bf16(f1.z, f1.w);
  ((uint4*)(emb_bf + wave * (EMB_DIM / 2)))[lane] = pk;

#pragma unroll
  for (int m = 16; m >= 1; m >>= 1) s += __shfl_xor(s, m, 32);
  if (lane == 0) enorm[wave] = s;
}

// ---------------------------------------------------------------------------
// Kernel 2: main VQ.  Per WG: 128 z-rows vs all 1024 codes via bf16 WMMA,
// argmin, gather of winning codebook rows, partial loss sum.
// ---------------------------------------------------------------------------
__global__ __launch_bounds__(THREADS) void vq_main(
    const float* __restrict__ z, const float* __restrict__ emb,
    const uint32_t* __restrict__ emb_bf, const float* __restrict__ enorm,
    float* __restrict__ out, float* __restrict__ partial) {
  // 64 KB: z tile as packed bf16 pairs, [row][128] uint32.
  __shared__ uint32_t zt[ROWS_PER_WG * (EMB_DIM / 2)];
  int*   rowIdx = (int*)zt;            // reused after A-fragments are loaded
  float* redBuf = (float*)(zt + ROWS_PER_WG);

  const int tid  = threadIdx.x;
  const int lane = tid & 31;
  const int wave = tid >> 5;           // 8 waves, 16 rows each
  const int m    = lane & 15;          // row/col within 16-slot group
  const int g16  = lane >> 4;          // half-wave select
  const int rowBase = blockIdx.x * ROWS_PER_WG;

  // ---- Stage z tile -> LDS as bf16 pairs (coalesced float2 loads) ----
  const float2* z2 = (const float2*)(z + (size_t)rowBase * EMB_DIM);
#pragma unroll
  for (int it = 0; it < (ROWS_PER_WG * EMB_DIM / 2) / THREADS; ++it) {
    int g = it * THREADS + tid;
    float2 f = z2[g];
    zt[g] = pack_bf16(f.x, f.y);
  }
  __syncthreads();

  // ---- Load this wave's A panel (16 rows x 256 depth) into 64 VGPRs ----
  // 16-bit A 16x32 layout: VGPR j holds K pair starting at
  // 2*(j&3) + 8*(lane>=16) + 16*(j>>2), rows striped over lanes 0..15.
  BF16Frag afrag[8];
  const int rowL = wave * 16 + m;
#pragma unroll
  for (int s = 0; s < 8; ++s) {
#pragma unroll
    for (int j = 0; j < 8; ++j) {
      int K = 32 * s + 2 * (j & 3) + 8 * g16 + 16 * (j >> 2);
      afrag[s].u[j] = zt[rowL * (EMB_DIM / 2) + (K >> 1)];
    }
  }
  __syncthreads();  // zt dead past here; rowIdx aliases it

  float minv[8];
  int   mini[8];
#pragma unroll
  for (int r = 0; r < 8; ++r) { minv[r] = 3.4e38f; mini[r] = 0; }

  // ---- Sweep 64 N-tiles of the codebook ----
  for (int t = 0; t < EMB_NUM / 16; ++t) {
    const int n = t * 16 + m;                         // this lane's column
    const uint32_t* brow = emb_bf + n * (EMB_DIM / 2);
    if (t + 1 < EMB_NUM / 16)
      __builtin_prefetch(brow + 16 * (EMB_DIM / 2), 0, 0);

    v8f acc = {};
#pragma unroll
    for (int s = 0; s < 8; ++s) {
      // 16-bit B 32x16 layout: VGPR j holds K pair
      // 2*(j&3) + 8*(lane>=16) + 16*(j>>2) (matches A): two b128 loads.
      BF16Frag bfrag;
      const uint4* b4 = (const uint4*)(brow + 16 * s + 4 * g16);
      *(uint4*)&bfrag.u[0] = b4[0];
      *(uint4*)&bfrag.u[4] = b4[2];
      acc = __builtin_amdgcn_wmma_f32_16x16x32_bf16(
          false, afrag[s].v, false, bfrag.v, (short)0, acc, false, false);
    }

    const float en = enorm[n];
#pragma unroll
    for (int r = 0; r < 8; ++r) {
      float dist = en - 2.0f * acc[r];   // ||e||^2 - 2 z.e  (||z||^2 const)
      if (dist < minv[r]) { minv[r] = dist; mini[r] = n; }
    }
  }

  // ---- Cross-lane argmin: C layout puts N across the 16-lane half,
  //      M = r (lanes 0-15) or r+8 (lanes 16-31).  xor masks 8..1 stay
  //      inside each half.  Tie-break on lower index (argmin semantics).
#pragma unroll
  for (int r = 0; r < 8; ++r) {
    float v = minv[r];
    int   i = mini[r];
#pragma unroll
    for (int msk = 8; msk >= 1; msk >>= 1) {
      float ov = __shfl_xor(v, msk, 32);
      int   oi = __shfl_xor(i, msk, 32);
      if (ov < v || (ov == v && oi < i)) { v = ov; i = oi; }
    }
    if (m == 0) rowIdx[wave * 16 + r + 8 * g16] = i;
  }
  __syncthreads();

  // ---- Gather winning rows -> out, accumulate (q - z)^2 (coalesced) ----
  float sum = 0.f;
  for (int k = 0; k < ROWS_PER_WG; ++k) {
    const int qi = rowIdx[k];
    const float q  = emb[qi * EMB_DIM + tid];
    const size_t o = (size_t)(rowBase + k) * EMB_DIM + tid;
    const float zv = z[o];
    out[o] = q;
    const float d = q - zv;
    sum += d * d;
  }

  // ---- Deterministic per-WG reduction ----
#pragma unroll
  for (int msk = 16; msk >= 1; msk >>= 1) sum += __shfl_xor(sum, msk, 32);
  if (lane == 0) redBuf[wave] = sum;
  __syncthreads();
  if (tid == 0) {
    float s = 0.f;
#pragma unroll
    for (int w = 0; w < THREADS / 32; ++w) s += redBuf[w];
    partial[blockIdx.x] = s;
  }
}

// ---------------------------------------------------------------------------
// Kernel 3: fixed-order reduction of 512 partials -> both scalar losses.
// ---------------------------------------------------------------------------
__global__ __launch_bounds__(256) void vq_loss(const float* __restrict__ partial,
                                               float* __restrict__ out) {
  __shared__ float buf[256];
  const int tid = threadIdx.x;
  buf[tid] = partial[tid] + partial[tid + 256];
  __syncthreads();
  for (int s = 128; s > 0; s >>= 1) {
    if (tid < s) buf[tid] += buf[tid + s];
    __syncthreads();
  }
  if (tid == 0) {
    const float loss = 0.5f * buf[0] / (float)((size_t)N_ROWS * EMB_DIM);
    out[(size_t)N_ROWS * EMB_DIM + 0] = loss;  // commitment_loss
    out[(size_t)N_ROWS * EMB_DIM + 1] = loss;  // emb_loss (numerically equal)
  }
}

extern "C" void kernel_launch(void* const* d_in, const int* in_sizes, int n_in,
                              void* d_out, int out_size, void* d_ws,
                              size_t ws_size, hipStream_t stream) {
  const float* z   = (const float*)d_in[0];
  const float* emb = (const float*)d_in[1];
  float* out = (float*)d_out;

  uint8_t* ws = (uint8_t*)d_ws;
  uint32_t* emb_bf = (uint32_t*)ws;                          // 512 KB
  float*    enorm  = (float*)(ws + 512 * 1024);              // 4 KB
  float*    part   = (float*)(ws + 512 * 1024 + 4 * 1024);   // 2 KB

  vq_prep<<<EMB_NUM / 8, 256, 0, stream>>>(emb, emb_bf, enorm);
  vq_main<<<NUM_WG, THREADS, 0, stream>>>(z, emb, emb_bf, enorm, out, part);
  vq_loss<<<1, 256, 0, stream>>>(part, out);
}
